// LSTM_4312147165798
// MI455X (gfx1250) — matrix operand order repaired and checked by reading, not compiled, not run on hardware
//
#include <hip/hip_runtime.h>
#include <hip/hip_bf16.h>

#define B_   32
#define T_   512
#define IN_  256
#define H_   1024
#define OUT_ 256
#define G4H  (4 * H_)

typedef __bf16 bf16_t;
typedef __attribute__((ext_vector_type(16))) __bf16 v16bf;
typedef __attribute__((ext_vector_type(8)))  float  v8f;

// ---------------------------------------------------------------------------
// helpers
// ---------------------------------------------------------------------------
__device__ __forceinline__ float fast_sigmoid(float x) {
  return 1.0f / (1.0f + __expf(-x));
}
__device__ __forceinline__ float fast_tanh(float x) {
  return 2.0f / (1.0f + __expf(-2.0f * x)) - 1.0f;
}

// A-fragment (16-bit A 16x32, MxK): lane L<16 -> row M=L, K = k0..k0+15
// lane L+16 -> row M=L, K = k0+16..k0+31. Caller pre-adds (hi?16:0).
__device__ __forceinline__ v16bf load_a_frag(const bf16_t* p) {
  v16bf out;
  ((uint4*)&out)[0] = *(const uint4*)(p);       // 8 bf16 at k
  ((uint4*)&out)[1] = *(const uint4*)(p + 8);   // 8 bf16 at k+8
  return out;
}

// B-fragment (16-bit B 32x16, KxN): lane = column N; low half K=0..15, high half
// K=16..31 (caller pre-adds (hi?16:0)); 16 contiguous bf16 from the weight row.
__device__ __forceinline__ v16bf load_b_frag(const bf16_t* p) {
  v16bf out;
  ((uint4*)&out)[0] = *(const uint4*)(p);
  ((uint4*)&out)[1] = *(const uint4*)(p + 8);
  return out;
}

__device__ __forceinline__ v8f wmma_bf16(v16bf a, v16bf b, v8f c) {
  return __builtin_amdgcn_wmma_f32_16x16x32_bf16(
      /*neg_a=*/false, a, /*neg_b=*/false, b,
      /*c_mod=*/(short)0, c, /*reuse_a=*/false, /*reuse_b=*/false);
}

// ---------------------------------------------------------------------------
// small utility kernels
// ---------------------------------------------------------------------------
__global__ void k_f32_to_bf16(const float* __restrict__ src,
                              bf16_t* __restrict__ dst, int n) {
  int i = blockIdx.x * blockDim.x + threadIdx.x;
  int stride = gridDim.x * blockDim.x;
  for (; i < n; i += stride) dst[i] = (bf16_t)src[i];
}

__global__ void k_bias_combine(const float* __restrict__ a,
                               const float* __restrict__ b,
                               float* __restrict__ o, int n) {
  int i = blockIdx.x * blockDim.x + threadIdx.x;
  if (i < n) o[i] = a[i] + b[i];
}

__global__ void k_zero_bf16(bf16_t* p, int n) {
  int i = blockIdx.x * blockDim.x + threadIdx.x;
  if (i < n) p[i] = (bf16_t)0.0f;
}

__global__ void k_zero_f32(float* p, int n) {
  int i = blockIdx.x * blockDim.x + threadIdx.x;
  if (i < n) p[i] = 0.0f;
}

// ---------------------------------------------------------------------------
// One LSTM timestep:
//   gates[b, j] = sum_k x_t[b,k]*Wih[j,k] + sum_k h[b,k]*Whh[j,k] + bias[j]
// grid = H/128 blocks, 256 threads (8 waves). Wave w owns 16 hidden columns
// (hcol = blockIdx.x*128 + w*16) and computes ALL four gates for them:
// 4 gates x 2 M-tiles (M=32 batch) of 16x16 f32 accumulators = 8 WMMA/k-chunk.
// The i,f,g,o combine then happens entirely in registers.
// h double-buffered (h_in != h_out) so cross-block RAW within a step is safe.
// Templated on KX / XLD so all loop trip counts are compile-time constants,
// and gate weight pointers are individual scalars so clang proves them
// global-addrspace (global_load_b128, not flat_load -> no DScnt coupling).
// ---------------------------------------------------------------------------
template <int KX, int XLD>
__global__ __launch_bounds__(256) void k_lstm_step(
    const bf16_t* __restrict__ xrow,  // + b*XLD = input row (KX feats) at time t
    const bf16_t* __restrict__ Wih,   // [4H, KX] row-major, bf16
    const bf16_t* __restrict__ Whh,   // [4H, H]  row-major, bf16
    const float*  __restrict__ bias,  // [4H]  (bih + bhh)
    const bf16_t* __restrict__ h_in,  // [B, H] bf16
    bf16_t* __restrict__ h_out,       // [B, H] bf16
    float*  __restrict__ c_state,     // [B, H] f32
    bf16_t* __restrict__ h_seq)       // + b*H : output slice [B][H] at time t
{
  const int tid  = threadIdx.x;
  const int wave = tid >> 5;
  const int lane = tid & 31;
  const int lh   = lane & 15;
  const int hi   = (lane >> 4) & 1;

  const int hcol = blockIdx.x * 128 + wave * 16;  // 16 hidden cols per wave

  v8f acc[4][2];
#pragma unroll
  for (int g = 0; g < 4; ++g)
#pragma unroll
    for (int m = 0; m < 2; ++m)
      acc[g][m] = (v8f){0.f, 0.f, 0.f, 0.f, 0.f, 0.f, 0.f, 0.f};

  // ---- input contribution: A rows = batch from xrow, B rows = Wih gate rows
  {
    const bf16_t* a0 = xrow + (long)(lh)      * XLD + (hi ? 16 : 0);
    const bf16_t* a1 = xrow + (long)(16 + lh) * XLD + (hi ? 16 : 0);
    const bf16_t* wp0 = Wih + (long)(0 * H_ + hcol + lh) * KX + (hi ? 16 : 0);
    const bf16_t* wp1 = Wih + (long)(1 * H_ + hcol + lh) * KX + (hi ? 16 : 0);
    const bf16_t* wp2 = Wih + (long)(2 * H_ + hcol + lh) * KX + (hi ? 16 : 0);
    const bf16_t* wp3 = Wih + (long)(3 * H_ + hcol + lh) * KX + (hi ? 16 : 0);
    for (int k = 0; k < KX; k += 32) {
      v16bf A0 = load_a_frag(a0);
      v16bf A1 = load_a_frag(a1);
      v16bf B0 = load_b_frag(wp0);
      v16bf B1 = load_b_frag(wp1);
      v16bf B2 = load_b_frag(wp2);
      v16bf B3 = load_b_frag(wp3);
      acc[0][0] = wmma_bf16(A0, B0, acc[0][0]);
      acc[0][1] = wmma_bf16(A1, B0, acc[0][1]);
      acc[1][0] = wmma_bf16(A0, B1, acc[1][0]);
      acc[1][1] = wmma_bf16(A1, B1, acc[1][1]);
      acc[2][0] = wmma_bf16(A0, B2, acc[2][0]);
      acc[2][1] = wmma_bf16(A1, B2, acc[2][1]);
      acc[3][0] = wmma_bf16(A0, B3, acc[3][0]);
      acc[3][1] = wmma_bf16(A1, B3, acc[3][1]);
      a0 += 32; a1 += 32;
      wp0 += 32; wp1 += 32; wp2 += 32; wp3 += 32;
    }
  }

  // ---- recurrent contribution: A rows = batch from h_in, B rows = Whh
  {
    const bf16_t* a0 = h_in + (long)(lh)      * H_ + (hi ? 16 : 0);
    const bf16_t* a1 = h_in + (long)(16 + lh) * H_ + (hi ? 16 : 0);
    const bf16_t* wp0 = Whh + (long)(0 * H_ + hcol + lh) * H_ + (hi ? 16 : 0);
    const bf16_t* wp1 = Whh + (long)(1 * H_ + hcol + lh) * H_ + (hi ? 16 : 0);
    const bf16_t* wp2 = Whh + (long)(2 * H_ + hcol + lh) * H_ + (hi ? 16 : 0);
    const bf16_t* wp3 = Whh + (long)(3 * H_ + hcol + lh) * H_ + (hi ? 16 : 0);
    for (int k = 0; k < H_; k += 32) {
      v16bf A0 = load_a_frag(a0);
      v16bf A1 = load_a_frag(a1);
      v16bf B0 = load_b_frag(wp0);
      v16bf B1 = load_b_frag(wp1);
      v16bf B2 = load_b_frag(wp2);
      v16bf B3 = load_b_frag(wp3);
      acc[0][0] = wmma_bf16(A0, B0, acc[0][0]);
      acc[0][1] = wmma_bf16(A1, B0, acc[0][1]);
      acc[1][0] = wmma_bf16(A0, B1, acc[1][0]);
      acc[1][1] = wmma_bf16(A1, B1, acc[1][1]);
      acc[2][0] = wmma_bf16(A0, B2, acc[2][0]);
      acc[2][1] = wmma_bf16(A1, B2, acc[2][1]);
      acc[3][0] = wmma_bf16(A0, B3, acc[3][0]);
      acc[3][1] = wmma_bf16(A1, B3, acc[3][1]);
      a0 += 32; a1 += 32;
      wp0 += 32; wp1 += 32; wp2 += 32; wp3 += 32;
    }
  }

  // ---- elementwise LSTM cell update, fully in registers.
  // C/D layout: VGPR r -> M = mt*16 + hi*8 + r, N = lh.
  const int n = hcol + lh;
  const float bi = bias[0 * H_ + n];
  const float bf = bias[1 * H_ + n];
  const float bg = bias[2 * H_ + n];
  const float bo = bias[3 * H_ + n];

#pragma unroll
  for (int mt = 0; mt < 2; ++mt) {
#pragma unroll
    for (int r = 0; r < 8; ++r) {
      const int  m   = mt * 16 + hi * 8 + r;  // batch index
      const long idx = (long)m * H_ + n;
      const float iv = fast_sigmoid(acc[0][mt][r] + bi);
      const float fv = fast_sigmoid(acc[1][mt][r] + bf);
      const float gv = fast_tanh   (acc[2][mt][r] + bg);
      const float ov = fast_sigmoid(acc[3][mt][r] + bo);
      const float c  = fv * c_state[idx] + iv * gv;
      c_state[idx]   = c;
      const float h  = ov * fast_tanh(c);
      const bf16_t hb = (bf16_t)h;
      h_out[idx] = hb;
      h_seq[idx] = hb;
    }
  }
}

// ---------------------------------------------------------------------------
// Final FC: out[b,t,o] = sum_h h2[t][b][h] * Wfc[o,h] + bfc[o]
// Flat row r = b*T + t. grid = (B*T/32, OUT/128), 256 threads (8 waves),
// wave w covers N-tile n_base = blockIdx.y*128 + w*16, M = 32 rows.
// ---------------------------------------------------------------------------
__global__ __launch_bounds__(256) void k_fc(
    const bf16_t* __restrict__ h2,   // [T][B][H] bf16
    const bf16_t* __restrict__ Wfc,  // [OUT][H] bf16
    const float*  __restrict__ bfc,  // [OUT]
    float* __restrict__ out)         // [B][T][OUT] f32
{
  const int tid  = threadIdx.x;
  const int wave = tid >> 5;
  const int lane = tid & 31;
  const int lh   = lane & 15;
  const int hi   = (lane >> 4) & 1;

  const int n_base = blockIdx.y * 128 + wave * 16;
  const int r0     = blockIdx.x * 32;  // 32 consecutive flat rows, same batch b
  const int b      = r0 >> 9;          // T = 512
  const int t0     = r0 & (T_ - 1);

  v8f acc[2];
#pragma unroll
  for (int m = 0; m < 2; ++m)
    acc[m] = (v8f){0.f, 0.f, 0.f, 0.f, 0.f, 0.f, 0.f, 0.f};

  const bf16_t* a0 = h2 + ((long)(t0 + lh)      * B_ + b) * H_ + (hi ? 16 : 0);
  const bf16_t* a1 = h2 + ((long)(t0 + 16 + lh) * B_ + b) * H_ + (hi ? 16 : 0);
  const bf16_t* wp = Wfc + (long)(n_base + lh) * H_ + (hi ? 16 : 0);

  for (int k = 0; k < H_; k += 32) {
    v16bf A0 = load_a_frag(a0);
    v16bf A1 = load_a_frag(a1);
    v16bf Bv = load_b_frag(wp);
    acc[0] = wmma_bf16(A0, Bv, acc[0]);
    acc[1] = wmma_bf16(A1, Bv, acc[1]);
    a0 += 32;
    a1 += 32;
    wp += 32;
  }

  const int   n  = n_base + lh;
  const float bv = bfc[n];
#pragma unroll
  for (int mt = 0; mt < 2; ++mt) {
#pragma unroll
    for (int r = 0; r < 8; ++r) {
      const int m   = mt * 16 + hi * 8 + r;
      const int row = r0 + m;
      out[(long)row * OUT_ + n] = acc[mt][r] + bv;
    }
  }
}

// ---------------------------------------------------------------------------
// host launcher
// ---------------------------------------------------------------------------
extern "C" void kernel_launch(void* const* d_in, const int* in_sizes, int n_in,
                              void* d_out, int out_size, void* d_ws, size_t ws_size,
                              hipStream_t stream) {
  (void)in_sizes; (void)n_in; (void)out_size; (void)ws_size;

  const float* x    = (const float*)d_in[0];
  const float* Wih0 = (const float*)d_in[1];
  const float* Whh0 = (const float*)d_in[2];
  const float* bih0 = (const float*)d_in[3];
  const float* bhh0 = (const float*)d_in[4];
  const float* Wih1 = (const float*)d_in[5];
  const float* Whh1 = (const float*)d_in[6];
  const float* bih1 = (const float*)d_in[7];
  const float* bhh1 = (const float*)d_in[8];
  const float* Wfc  = (const float*)d_in[9];
  const float* bfc  = (const float*)d_in[10];
  float* out = (float*)d_out;

  char*  ws  = (char*)d_ws;
  size_t off = 0;
  auto alloc = [&](size_t bytes) -> void* {
    void* p = ws + off;
    off += (bytes + 255) & ~(size_t)255;
    return p;
  };

  bf16_t* xb    = (bf16_t*)alloc((size_t)B_ * T_ * IN_ * sizeof(bf16_t));
  bf16_t* wih0  = (bf16_t*)alloc((size_t)G4H * IN_ * sizeof(bf16_t));
  bf16_t* whh0  = (bf16_t*)alloc((size_t)G4H * H_ * sizeof(bf16_t));
  bf16_t* wih1  = (bf16_t*)alloc((size_t)G4H * H_ * sizeof(bf16_t));
  bf16_t* whh1  = (bf16_t*)alloc((size_t)G4H * H_ * sizeof(bf16_t));
  bf16_t* wfc   = (bf16_t*)alloc((size_t)OUT_ * H_ * sizeof(bf16_t));
  float*  bias0 = (float*) alloc((size_t)G4H * sizeof(float));
  float*  bias1 = (float*) alloc((size_t)G4H * sizeof(float));
  bf16_t* hA    = (bf16_t*)alloc((size_t)B_ * H_ * sizeof(bf16_t));
  bf16_t* hB    = (bf16_t*)alloc((size_t)B_ * H_ * sizeof(bf16_t));
  float*  cS    = (float*) alloc((size_t)B_ * H_ * sizeof(float));
  bf16_t* h1seq = (bf16_t*)alloc((size_t)T_ * B_ * H_ * sizeof(bf16_t));
  bf16_t* h2seq = (bf16_t*)alloc((size_t)T_ * B_ * H_ * sizeof(bf16_t));

  // ---- precision conversion + bias fold (one-time per call) ----
  k_f32_to_bf16<<<dim3(1024), 256, 0, stream>>>(x,    xb,   B_ * T_ * IN_);
  k_f32_to_bf16<<<dim3(1024), 256, 0, stream>>>(Wih0, wih0, G4H * IN_);
  k_f32_to_bf16<<<dim3(1024), 256, 0, stream>>>(Whh0, whh0, G4H * H_);
  k_f32_to_bf16<<<dim3(1024), 256, 0, stream>>>(Wih1, wih1, G4H * H_);
  k_f32_to_bf16<<<dim3(1024), 256, 0, stream>>>(Whh1, whh1, G4H * H_);
  k_f32_to_bf16<<<dim3(1024), 256, 0, stream>>>(Wfc,  wfc,  OUT_ * H_);
  k_bias_combine<<<dim3((G4H + 255) / 256), 256, 0, stream>>>(bih0, bhh0, bias0, G4H);
  k_bias_combine<<<dim3((G4H + 255) / 256), 256, 0, stream>>>(bih1, bhh1, bias1, G4H);

  const int nState = B_ * H_;
  const dim3 zgrid((nState + 255) / 256);

  // ---- layer 0 ----
  k_zero_bf16<<<zgrid, 256, 0, stream>>>(hA, nState);
  k_zero_f32 <<<zgrid, 256, 0, stream>>>(cS, nState);
  for (int t = 0; t < T_; ++t) {
    const bf16_t* hin  = (t & 1) ? hB : hA;
    bf16_t*       hout = (t & 1) ? hA : hB;
    k_lstm_step<IN_, T_ * IN_><<<dim3(H_ / 128), 256, 0, stream>>>(
        xb + (size_t)t * IN_, wih0, whh0, bias0,
        hin, hout, cS, h1seq + (size_t)t * B_ * H_);
  }

  // ---- layer 1 ----
  k_zero_bf16<<<zgrid, 256, 0, stream>>>(hA, nState);
  k_zero_f32 <<<zgrid, 256, 0, stream>>>(cS, nState);
  for (int t = 0; t < T_; ++t) {
    const bf16_t* hin  = (t & 1) ? hB : hA;
    bf16_t*       hout = (t & 1) ? hA : hB;
    k_lstm_step<H_, H_><<<dim3(H_ / 128), 256, 0, stream>>>(
        h1seq + (size_t)t * B_ * H_, wih1, whh1, bias1,
        hin, hout, cS, h2seq + (size_t)t * B_ * H_);
  }

  // ---- final FC ----
  k_fc<<<dim3(B_ * T_ / 32, OUT_ / 128), 256, 0, stream>>>(h2seq, wfc, bfc, out);
}
